// SparseCodingLayer_First_ReLU_23270132809948
// MI455X (gfx1250) — matrix operand order repaired and checked by reading, not compiled, not run on hardware
//
#include <hip/hip_runtime.h>

typedef __attribute__((ext_vector_type(16))) __bf16 v16bf;
typedef __attribute__((ext_vector_type(8)))  float  v8f;

#define CIN 64
#define FS  512
#define OD  128
#define HH  112
#define WW  112
#define HO  110
#define WO  110
#define CK  576   // 64*9

#define ENC_ELEMS (FS*CK)   // 294912 bf16 fragment elems
#define DEC_ELEMS (CK*FS)   // 294912
#define RED_ELEMS (OD*FS)   // 65536

// ---------------------------------------------------------------------------
// Rewrite f32 weights into bf16 WMMA B-operand fragment order.
// B 32x16 (KxN) layout (wave32): lane<16 -> col=lane, K={0..7,16..23};
// lane>=16 -> col=lane-16, K={8..15,24..31}.  Fragment = 32 lanes x 16 bf16 contiguous.
// ---------------------------------------------------------------------------
__global__ void prep_weights_kernel(const float* __restrict__ enc_w,
                                    const float* __restrict__ dec_w,
                                    const float* __restrict__ red_w,
                                    unsigned short* __restrict__ wsu) {
  __bf16* dst = (__bf16*)wsu;
  int e = blockIdx.x * 256 + threadIdx.x;
  if (e < ENC_ELEMS) {
    // enc: K = 576 (ordering: ks = (kh*3+kw)*2 + cb, c = cb*32 + Klocal), N = 512
    int j = e & 15, lane = (e >> 4) & 31, nt = (e >> 9) & 31, ks = e >> 14; // ks 0..17
    int n = nt * 16 + (lane & 15);
    int K = ((lane & 16) ? 8 : 0) + (j & 7) + ((j >= 8) ? 16 : 0);
    int off = ks >> 1, cb = ks & 1;
    int c = cb * 32 + K;
    dst[e] = (__bf16)enc_w[n * CK + c * 9 + off];
  } else if (e < ENC_ELEMS + DEC_ELEMS) {
    // dec: K = 512, N = 576 (36 n-tiles)
    int i = e - ENC_ELEMS;
    int j = i & 15, lane = (i >> 4) & 31, t = i >> 9;   // t 0..575
    int nt = t % 36, ks = t / 36;
    int n  = nt * 16 + (lane & 15);
    int kk = ks * 32 + ((lane & 16) ? 8 : 0) + (j & 7) + ((j >= 8) ? 16 : 0);
    dst[e] = (__bf16)dec_w[n * FS + kk];
  } else if (e < ENC_ELEMS + DEC_ELEMS + RED_ELEMS) {
    // red: K = 512, N = 128 (8 n-tiles)
    int i = e - ENC_ELEMS - DEC_ELEMS;
    int j = i & 15, lane = (i >> 4) & 31, t = i >> 9;   // t 0..127
    int nt = t & 7, ks = t >> 3;
    int n  = nt * 16 + (lane & 15);
    int kk = ks * 32 + ((lane & 16) ? 8 : 0) + (j & 7) + ((j >= 8) ? 16 : 0);
    dst[e] = (__bf16)red_w[n * FS + kk];
  }
}

// ---------------------------------------------------------------------------
// Fused: enc-conv + ReLU  ->  dec-conv + sigmoid + MSE partials  ->  red-conv
// Block = 256 threads = 8 waves as 2(M) x 4(N); M-tile = 32 positions along w.
// ---------------------------------------------------------------------------
__global__ __launch_bounds__(256)
void sparse_coding_fused_kernel(const float* __restrict__ x,
                                const float* __restrict__ enc_b,
                                const float* __restrict__ dec_b,
                                const float* __restrict__ red_b,
                                const unsigned short* __restrict__ encFu,
                                const unsigned short* __restrict__ decFu,
                                const unsigned short* __restrict__ redFu,
                                float* __restrict__ out,
                                float* __restrict__ partials) {
  // 18 ksteps x 2 mtiles x 32 lanes x 16 bf16 = 36864 B; reused for h frags
  __shared__ __align__(32) __bf16 sFrag[18432];
  __shared__ float sRed[8];

  const int tid  = threadIdx.x;
  const int lane = tid & 31;
  const int wave = tid >> 5;
  const int wm   = wave & 1;       // 16-row half
  const int wn   = wave >> 1;      // 128-col group (phase 1)
  const int bid  = blockIdx.x;
  const int wstrip = bid & 3;
  const int hh   = (bid >> 2) % HO;
  const int bb   = bid / (4 * HO);
  const int w0   = wstrip * 32;

  const v16bf* encF = (const v16bf*)encFu;
  const v16bf* decF = (const v16bf*)decFu;
  const v16bf* redF = (const v16bf*)redFu;
  const v16bf* aBase = (const v16bf*)sFrag;
  const v8f zero8 = {0.f,0.f,0.f,0.f,0.f,0.f,0.f,0.f};

  // ---- stage im2col A-fragments (WMMA A layout: lane<16 Ks {0..7,16..23},
  //      lane>=16 Ks {8..15,24..31}; row = lane&15) -------------------------
  for (int i = tid; i < 18432; i += 256) {
    int lane2 = i & 31;
    int j   = (i >> 5) & 15;
    int fid = i >> 9;                         // = ks*2 + mt, 0..35
    int mt = fid & 1, ks = fid >> 1;
    int off = ks >> 1, cb = ks & 1;
    int K = (j & 7) + ((j >= 8) ? 16 : 0) + ((lane2 >= 16) ? 8 : 0);
    int c = cb * 32 + K;
    int r = mt * 16 + (lane2 & 15);
    int w = w0 + r; w = (w > WO - 1) ? (WO - 1) : w;   // clamp tail
    int kh = off / 3, kw = off - kh * 3;
    float v = x[((bb * CIN + c) * HH + (hh + kh)) * WW + (w + kw)];
    sFrag[(fid * 32 + lane2) * 16 + j] = (__bf16)v;
  }
  __syncthreads();

  // ---- Phase 1: h = relu(A * encW^T + b), h tile 32x512 in accumulators ---
  v8f acc[8];
#pragma unroll
  for (int t = 0; t < 8; ++t) acc[t] = zero8;

  for (int ks = 0; ks < 18; ++ks) {
    v16bf a = aBase[(ks * 2 + wm) * 32 + lane];
#pragma unroll
    for (int nt = 0; nt < 8; ++nt) {
      v16bf b = encF[(ks * 32 + (wn * 8 + nt)) * 32 + lane];
      acc[nt] = __builtin_amdgcn_wmma_f32_16x16x32_bf16(
          false, a, false, b, (short)0, acc[nt], false, false);
    }
  }
  __syncthreads();   // everyone done reading A-fragments

  // ---- bias + ReLU, scatter h into LDS in A-fragment layout (K = fs) ------
  const int rowAdd = (lane & 16) ? 8 : 0;
#pragma unroll
  for (int nt = 0; nt < 8; ++nt) {
    int n  = wn * 128 + nt * 16 + (lane & 15);
    float bv = enc_b[n];
    int ksd = n >> 5, kl = n & 31;
    int jj = (kl & 7) + ((kl & 16) ? 8 : 0);
    int laneBase = (kl & 8) ? 16 : 0;
#pragma unroll
    for (int e = 0; e < 8; ++e) {
      float hv = acc[nt][e] + bv;
      hv = hv > 0.f ? hv : 0.f;
      int r = wm * 16 + rowAdd + e;
      int lane2 = (r & 15) + laneBase;
      sFrag[((ksd * 2 + (r >> 4)) * 32 + lane2) * 16 + jj] = (__bf16)hv;
    }
  }
  __syncthreads();

  // ---- Phase 2a: aux = sigmoid(h*decW^T + b), MSE vs im2col(x) ------------
  float lsum = 0.f;
  for (int nti = 0; nti < 9; ++nti) {
    int nt = wn + nti * 4;                  // 0..35
    v8f a3 = zero8;
    for (int ks = 0; ks < 16; ++ks) {
      v16bf hA = aBase[(ks * 2 + wm) * 32 + lane];
      v16bf bB = decF[(ks * 36 + nt) * 32 + lane];
      a3 = __builtin_amdgcn_wmma_f32_16x16x32_bf16(
          false, hA, false, bB, (short)0, a3, false, false);
    }
    int n2 = nt * 16 + (lane & 15);
    float bv = dec_b[n2];
    int c = n2 / 9, off = n2 - c * 9;
    int kh = off / 3, kw = off - kh * 3;
    const float* xs = x + ((bb * CIN + c) * HH + (hh + kh)) * WW + kw;
#pragma unroll
    for (int e = 0; e < 8; ++e) {
      int w = w0 + wm * 16 + rowAdd + e;
      if (w < WO) {
        float t  = a3[e] + bv;
        float sg = 1.f / (1.f + __expf(-t));
        float d  = xs[w] - sg;
        lsum += d * d;
      }
    }
  }

  // ---- Phase 2b: out = h*redW^T + b ---------------------------------------
  for (int nti = 0; nti < 2; ++nti) {
    int nt = wn * 2 + nti;                  // 0..7
    v8f a4 = zero8;
    for (int ks = 0; ks < 16; ++ks) {
      v16bf hA = aBase[(ks * 2 + wm) * 32 + lane];
      v16bf bB = redF[(ks * 8 + nt) * 32 + lane];
      a4 = __builtin_amdgcn_wmma_f32_16x16x32_bf16(
          false, hA, false, bB, (short)0, a4, false, false);
    }
    int n = nt * 16 + (lane & 15);
    float bv = red_b[n];
    float* op = out + ((bb * OD + n) * HO + hh) * WO;
#pragma unroll
    for (int e = 0; e < 8; ++e) {
      int w = w0 + wm * 16 + rowAdd + e;
      if (w < WO) op[w] = a4[e] + bv;
    }
  }

  // ---- block loss reduction (deterministic, no float atomics) -------------
  lsum += __shfl_xor(lsum, 16);
  lsum += __shfl_xor(lsum, 8);
  lsum += __shfl_xor(lsum, 4);
  lsum += __shfl_xor(lsum, 2);
  lsum += __shfl_xor(lsum, 1);
  if (lane == 0) sRed[wave] = lsum;
  __syncthreads();
  if (tid == 0) {
    float s = 0.f;
#pragma unroll
    for (int i = 0; i < 8; ++i) s += sRed[i];
    partials[bid] = s;
  }
}

// ---------------------------------------------------------------------------
__global__ void loss_reduce_kernel(const float* __restrict__ partials, int n,
                                   float* __restrict__ dst, float scale) {
  __shared__ float s[256];
  float v = 0.f;
  for (int i = threadIdx.x; i < n; i += 256) v += partials[i];
  s[threadIdx.x] = v;
  __syncthreads();
  for (int o = 128; o > 0; o >>= 1) {
    if (threadIdx.x < o) s[threadIdx.x] += s[threadIdx.x + o];
    __syncthreads();
  }
  if (threadIdx.x == 0) dst[0] = s[0] * scale;
}

// ---------------------------------------------------------------------------
extern "C" void kernel_launch(void* const* d_in, const int* in_sizes, int n_in,
                              void* d_out, int out_size, void* d_ws, size_t ws_size,
                              hipStream_t stream) {
  const float* x     = (const float*)d_in[0];
  const float* enc_w = (const float*)d_in[1];
  const float* enc_b = (const float*)d_in[2];
  const float* dec_w = (const float*)d_in[3];
  const float* dec_b = (const float*)d_in[4];
  const float* red_w = (const float*)d_in[5];
  const float* red_b = (const float*)d_in[6];
  float* out = (float*)d_out;

  unsigned short* wsu  = (unsigned short*)d_ws;
  unsigned short* encF = wsu;
  unsigned short* decF = encF + ENC_ELEMS;
  unsigned short* redF = decF + DEC_ELEMS;
  float* partials = (float*)(redF + RED_ELEMS);

  const int totalW = ENC_ELEMS + DEC_ELEMS + RED_ELEMS;   // 655360
  prep_weights_kernel<<<(totalW + 255) / 256, 256, 0, stream>>>(enc_w, dec_w, red_w, wsu);

  const int nblocks = 8 * HO * 4;   // 3520
  sparse_coding_fused_kernel<<<nblocks, 256, 0, stream>>>(
      x, enc_b, dec_b, red_b, encF, decF, redF, out, partials);

  const float scale = 1.0f / (8.0f * 576.0f * 110.0f * 110.0f);
  loss_reduce_kernel<<<1, 256, 0, stream>>>(partials, nblocks, out + (out_size - 1), scale);
}
